// QuantizedLinear_35820027249033
// MI455X (gfx1250) — matrix operand order repaired and checked by reading, not compiled, not run on hardware
//
#include <hip/hip_runtime.h>

// ---------------------------------------------------------------------------
// QuantizedLinear for MI455X (gfx1250, wave32, WMMA, async-to-LDS)
//   y = x @ quantize(w).T + bias,  quantize(w) = clamp(ceil(w-0.5), -4, 3)
//
// Pass 0: w f32 -> quantized f16 in ws   (levels exact in f16)
// Pass 1: x f32 -> f16 in ws             (removes all cvt from GEMM loop)
// Pass 2: 128x128x64 double-buffered WMMA GEMM; tiles staged with
//         global_load_async_to_lds_b128 (ASYNCcnt), 4 waves x (64x64 C tile),
//         32x v_wmma_f32_16x16x32_f16 per K-stage per wave.
// Fallback (small ws): round-1 style kernel with on-the-fly x conversion.
// ---------------------------------------------------------------------------

typedef __attribute__((ext_vector_type(16))) _Float16 v16h;
typedef __attribute__((ext_vector_type(8)))  _Float16 v8h;
typedef __attribute__((ext_vector_type(4)))  _Float16 v4h;
typedef __attribute__((ext_vector_type(8)))  float    v8f;
typedef __attribute__((ext_vector_type(4)))  float    v4f;
typedef __attribute__((ext_vector_type(4)))  int      v4i;

#define BM 128
#define BN 128
#define BK 64
#define PAD 8                  // halves
#define LSTR (BK + PAD)        // 72 halves = 144 B row stride (16B aligned)
#define ABUF (BM * LSTR)       // halves per A buffer

// ---------------- CDNA5 async-copy + wait helpers --------------------------
typedef __attribute__((address_space(1))) v4i  g_v4i;   // global int4
typedef __attribute__((address_space(3))) v4i  l_v4i;   // LDS int4
typedef __attribute__((address_space(3))) void lds_void;

__device__ __forceinline__ void async_copy_b128(const _Float16* g, _Float16* l) {
#if defined(__has_builtin) && __has_builtin(__builtin_amdgcn_global_load_async_to_lds_b128)
    __builtin_amdgcn_global_load_async_to_lds_b128(
        (g_v4i*)g, (l_v4i*)l, /*offset=*/0, /*cpol=*/0);
#else
    asm volatile("global_load_async_to_lds_b128 %0, %1, off"
                 :: "v"((unsigned)(uintptr_t)(lds_void*)l), "v"(g)
                 : "memory");
#endif
}

template <int N>
__device__ __forceinline__ void wait_asynccnt() {
#if defined(__has_builtin) && __has_builtin(__builtin_amdgcn_s_wait_asynccnt)
    __builtin_amdgcn_s_wait_asynccnt((unsigned short)N);
#else
    asm volatile("s_wait_asynccnt %0" :: "i"(N) : "memory");
#endif
}

// ---------------- Pass 0: quantize weights f32 -> f16 ----------------------
__device__ __forceinline__ _Float16 quant_level(float w) {
    float t = __builtin_ceilf(w - 0.5f);
    t = fminf(fmaxf(t, -4.0f), 3.0f);
    return (_Float16)t;
}

__global__ void quantize_weights_kernel(const float* __restrict__ w,
                                        _Float16* __restrict__ wq, int n) {
    int i = (blockIdx.x * blockDim.x + threadIdx.x) * 4;
    if (i + 3 >= n) return;                   // n is a multiple of 1024
    v4f v = *(const v4f*)(w + i);
    v4h q;
    q.x = quant_level(v.x); q.y = quant_level(v.y);
    q.z = quant_level(v.z); q.w = quant_level(v.w);
    *(v4h*)(wq + i) = q;
}

// ---------------- Pass 1: convert x f32 -> f16 -----------------------------
__global__ void convert_x_kernel(const float* __restrict__ x,
                                 _Float16* __restrict__ xh, int n) {
    int i = (blockIdx.x * blockDim.x + threadIdx.x) * 4;
    if (i + 3 >= n) return;
    v4f v = *(const v4f*)(x + i);
    v4h h = { (_Float16)v.x, (_Float16)v.y, (_Float16)v.z, (_Float16)v.w };
    *(v4h*)(xh + i) = h;
}

// ---------------- Pass 2: async double-buffered WMMA GEMM ------------------
// block = 128 threads = 4 wave32s in a 2x2 grid, each wave owns 64x64 of C.
__global__ __launch_bounds__(128) void gemm_wmma_async_kernel(
    const _Float16* __restrict__ xh,    // [M,K] f16
    const _Float16* __restrict__ wq,    // [N,K] f16 (quantized)
    const float*    __restrict__ bias,  // [N]
    float*          __restrict__ out,   // [M,N]
    int M, int N, int K) {

    __shared__ _Float16 ldsA[2 * ABUF];   // 36 KB
    __shared__ _Float16 ldsB[2 * ABUF];   // 36 KB

    const int tid    = threadIdx.x;
    const int lane   = tid & 31;
    const int wave   = tid >> 5;          // 0..3
    const int wm     = wave & 1;          // M offset wm*64
    const int wn     = wave >> 1;         // N offset wn*64
    const int laneLo = lane & 15;
    const int laneHi = lane >> 4;

    const int m0 = blockIdx.y * BM;
    const int n0 = blockIdx.x * BN;       // x over N -> A-slab L2 reuse

    // per-thread staging assignment: 8 chunks of 16B for A and for B.
    // chunk c = tid + i*128 -> row = (tid>>3) + 16*i, seg = (tid&7)*8 halves.
    const int sRow = tid >> 3;            // 0..15
    const int sSeg = (tid & 7) * 8;       // halves, 16B units
    const _Float16* gA = xh + (size_t)(m0 + sRow) * K + sSeg;
    const _Float16* gB = wq + (size_t)(n0 + sRow) * K + sSeg;
    _Float16* lA = ldsA + sRow * LSTR + sSeg;
    _Float16* lB = ldsB + sRow * LSTR + sSeg;

    v8f acc[4][4] = {};

    const int numK = K / BK;

    // prologue: stage k-block 0 into buffer 0 (16 async instrs / thread)
#pragma unroll
    for (int i = 0; i < 8; ++i) {
        async_copy_b128(gA + (size_t)16 * i * K, lA + 16 * i * LSTR);
        async_copy_b128(gB + (size_t)16 * i * K, lB + 16 * i * LSTR);
    }

    int buf = 0;
    for (int ks = 0; ks < numK; ++ks) {
        // all waves finished READING buf^1 (in iteration ks-1) before refill
        __syncthreads();

        if (ks + 1 < numK) {
            const size_t koff = (size_t)(ks + 1) * BK;
            _Float16* dA = lA + (buf ^ 1) * ABUF;
            _Float16* dB = lB + (buf ^ 1) * ABUF;
#pragma unroll
            for (int i = 0; i < 8; ++i) {
                async_copy_b128(gA + koff + (size_t)16 * i * K, dA + 16 * i * LSTR);
                async_copy_b128(gB + koff + (size_t)16 * i * K, dB + 16 * i * LSTR);
            }
            wait_asynccnt<16>();   // the 16 copies of the CURRENT stage are done
        } else {
            wait_asynccnt<0>();
        }
        __syncthreads();           // current buffer visible from all waves

        const _Float16* bA = ldsA + buf * ABUF;
        const _Float16* bB = ldsB + buf * ABUF;

#pragma unroll
        for (int kh = 0; kh < 2; ++kh) {   // two 32-K halves of the 64-K stage
            // A fragments: 16-bit A(16x32) layout.
            //  lane<16 : M=lane, K 0..7 -> v0..3, K16..23 -> v4..7
            //  lane>=16: M=lane-16, K 8..15 -> v0..3, K24..31 -> v4..7
            v16h aF[4];
#pragma unroll
            for (int mf = 0; mf < 4; ++mf) {
                const _Float16* p = bA + (wm * 64 + mf * 16 + laneLo) * LSTR + kh * 32;
                *(v8h*)&aF[mf]         = *(const v8h*)(p + laneHi * 8);
                *(((v8h*)&aF[mf]) + 1) = *(const v8h*)(p + 16 + laneHi * 8);
            }
            // B fragments: B(32x16): lane<16: N=lane,K0..7,K16..23; lane>=16: N=lane-16,K8..15,K24..31
            v16h bF[4];
#pragma unroll
            for (int nf = 0; nf < 4; ++nf) {
                const _Float16* p = bB + (wn * 64 + nf * 16 + laneLo) * LSTR + kh * 32;
                *(v8h*)&bF[nf]         = *(const v8h*)(p + laneHi * 8);
                *(((v8h*)&bF[nf]) + 1) = *(const v8h*)(p + 16 + laneHi * 8);
            }
#pragma unroll
            for (int mf = 0; mf < 4; ++mf)
#pragma unroll
                for (int nf = 0; nf < 4; ++nf)
                    acc[mf][nf] = __builtin_amdgcn_wmma_f32_16x16x32_f16(
                        false, aF[mf], false, bF[nf],
                        (short)0, acc[mf][nf], false, false);
        }
        buf ^= 1;
    }

    // epilogue: VGPR r of a tile holds M = r + (lane>=16)*8, N = lane%16
#pragma unroll
    for (int mf = 0; mf < 4; ++mf) {
#pragma unroll
        for (int nf = 0; nf < 4; ++nf) {
            const int col     = n0 + wn * 64 + nf * 16 + laneLo;
            const float bv    = bias[col];
            const int rowBase = m0 + wm * 64 + mf * 16 + laneHi * 8;
            float* op = out + (size_t)rowBase * N + col;
#pragma unroll
            for (int r = 0; r < 8; ++r) { *op = acc[mf][nf][r] + bv; op += N; }
        }
    }
}

// ---------------- Fallback GEMM (small workspace): on-the-fly x cvt --------
#define FTK 32
#define FSTR (FTK + PAD)
__global__ __launch_bounds__(256) void gemm_wmma_fallback_kernel(
    const float* __restrict__ x, const _Float16* __restrict__ wq,
    const float* __restrict__ bias, float* __restrict__ out,
    int M, int N, int K) {

    __shared__ _Float16 ldsA[BM * FSTR];
    __shared__ _Float16 ldsB[BN * FSTR];

    const int tid = threadIdx.x, lane = tid & 31, wave = tid >> 5;
    const int wm = wave & 1, wn = wave >> 1;
    const int laneLo = lane & 15, laneHi = lane >> 4;
    const int m0 = blockIdx.y * BM, n0 = blockIdx.x * BN;
    const int ldRow = tid >> 1, ldSeg = (tid & 1) * 16;

    const float*    xg = x  + (size_t)(m0 + ldRow) * K + ldSeg;
    const _Float16* bg = wq + (size_t)(n0 + ldRow) * K + ldSeg;

    v8f acc[4][2] = {};
    for (int k0 = 0; k0 < K; k0 += FTK) {
        v4f a0 = *(const v4f*)(xg + k0),     a1 = *(const v4f*)(xg + k0 + 4);
        v4f a2 = *(const v4f*)(xg + k0 + 8), a3 = *(const v4f*)(xg + k0 + 12);
        v8h b0 = *(const v8h*)(bg + k0),     b1 = *(const v8h*)(bg + k0 + 8);
        __syncthreads();
        v8h p0 = { (_Float16)a0.x,(_Float16)a0.y,(_Float16)a0.z,(_Float16)a0.w,
                   (_Float16)a1.x,(_Float16)a1.y,(_Float16)a1.z,(_Float16)a1.w };
        v8h p1 = { (_Float16)a2.x,(_Float16)a2.y,(_Float16)a2.z,(_Float16)a2.w,
                   (_Float16)a3.x,(_Float16)a3.y,(_Float16)a3.z,(_Float16)a3.w };
        *(v8h*)&ldsA[ldRow * FSTR + ldSeg]     = p0;
        *(v8h*)&ldsA[ldRow * FSTR + ldSeg + 8] = p1;
        *(v8h*)&ldsB[ldRow * FSTR + ldSeg]     = b0;
        *(v8h*)&ldsB[ldRow * FSTR + ldSeg + 8] = b1;
        __syncthreads();
        v16h aF[4], bF[2];
#pragma unroll
        for (int mf = 0; mf < 4; ++mf) {
            const _Float16* p = &ldsA[(wm * 64 + mf * 16 + laneLo) * FSTR];
            *(v8h*)&aF[mf]         = *(const v8h*)(p + laneHi * 8);
            *(((v8h*)&aF[mf]) + 1) = *(const v8h*)(p + 16 + laneHi * 8);
        }
#pragma unroll
        for (int nf = 0; nf < 2; ++nf) {
            const _Float16* p = &ldsB[(wn * 32 + nf * 16 + laneLo) * FSTR];
            *(v8h*)&bF[nf]         = *(const v8h*)(p + laneHi * 8);
            *(((v8h*)&bF[nf]) + 1) = *(const v8h*)(p + 16 + laneHi * 8);
        }
#pragma unroll
        for (int mf = 0; mf < 4; ++mf)
#pragma unroll
            for (int nf = 0; nf < 2; ++nf)
                acc[mf][nf] = __builtin_amdgcn_wmma_f32_16x16x32_f16(
                    false, aF[mf], false, bF[nf], (short)0, acc[mf][nf], false, false);
    }
#pragma unroll
    for (int mf = 0; mf < 4; ++mf)
#pragma unroll
        for (int nf = 0; nf < 2; ++nf) {
            const int col = n0 + wn * 32 + nf * 16 + laneLo;
            const float bv = bias[col];
            const int rowBase = m0 + wm * 64 + mf * 16 + laneHi * 8;
            float* op = out + (size_t)rowBase * N + col;
#pragma unroll
            for (int r = 0; r < 8; ++r) { *op = acc[mf][nf][r] + bv; op += N; }
        }
}

// ---------------------------------------------------------------------------
extern "C" void kernel_launch(void* const* d_in, const int* in_sizes, int n_in,
                              void* d_out, int out_size, void* d_ws, size_t ws_size,
                              hipStream_t stream) {
    const float* x    = (const float*)d_in[0];
    const float* w    = (const float*)d_in[1];
    const float* bias = (const float*)d_in[2];
    float* out        = (float*)d_out;

    const int N = in_sizes[2];
    const int K = in_sizes[1] / N;
    const int M = in_sizes[0] / K;

    _Float16* wq = (_Float16*)d_ws;                       // N*K halves
    const size_t wqElems = (size_t)N * K;
    const size_t xhElems = (size_t)M * K;

    quantize_weights_kernel<<<(int)(wqElems / 4 / 256), 256, 0, stream>>>(
        w, wq, (int)wqElems);

    if (ws_size >= (wqElems + xhElems) * sizeof(_Float16)) {
        _Float16* xh = wq + wqElems;
        convert_x_kernel<<<(int)(xhElems / 4 / 256), 256, 0, stream>>>(
            x, xh, (int)xhElems);
        dim3 grid(N / BN, M / BM);
        gemm_wmma_async_kernel<<<grid, 128, 0, stream>>>(xh, wq, bias, out, M, N, K);
    } else {
        dim3 grid(N / BN, M / BM);
        gemm_wmma_fallback_kernel<<<grid, 256, 0, stream>>>(x, wq, bias, out, M, N, K);
    }
}
